// TransformerBlock_42554535969110
// MI455X (gfx1250) — matrix operand order repaired and checked by reading, not compiled, not run on hardware
//
#include <hip/hip_runtime.h>
#include <hip/hip_bf16.h>

// ---------------- problem constants ----------------
constexpr int Bc  = 2;
constexpr int Lc  = 2048;
constexpr int Dc  = 1024;
constexpr int NHc = 16;
constexpr int HDc = 64;
constexpr int LATc= 16;
constexpr int Ec  = 8;
constexpr int Sc  = 2;
constexpr int HFFc= 4096;
constexpr int Tc  = Bc * Lc;     // 4096 tokens

typedef __attribute__((ext_vector_type(16))) __bf16 v16bf;
typedef __attribute__((ext_vector_type(8)))  __bf16 v8bf;
typedef __attribute__((ext_vector_type(8)))  float  v8f;

union FragA { v8bf h[2]; v16bf v; };
union FragB { v8bf h[2]; v16bf v; };

__device__ __forceinline__ float gelu_f(float x) {
    return 0.5f * x * (1.0f + erff(x * 0.70710678118654752f));
}

// ---- CDNA5 async copy: global -> LDS, tracked with ASYNCcnt ----
// (inline asm: portable across ROCm7.2 / amdgpu-toolchain, per CDNA5_HIP.md)
__device__ __forceinline__ void async_load16(const void* gptr, void* sptr) {
    unsigned lds = (unsigned)(uintptr_t)sptr;   // low 32 bits of flat = LDS offset
    asm volatile("global_load_async_to_lds_b128 %0, %1, off"
                 :: "v"(lds), "v"(gptr) : "memory");
}
__device__ __forceinline__ void wait_async0() {
    asm volatile("s_wait_asynccnt 0x0" ::: "memory");
}

// ---------------- LayerNorm -> bf16 ----------------
__global__ __launch_bounds__(256)
void ln_kernel(const float* __restrict__ x, const float* __restrict__ g,
               const float* __restrict__ b, __bf16* __restrict__ out)
{
    int t = blockIdx.x;
    const float* xr = x + (size_t)t * Dc;
    __shared__ float red[256];
    int tid = threadIdx.x;
    float s = 0.f;
    for (int i = tid; i < Dc; i += 256) s += xr[i];
    red[tid] = s; __syncthreads();
    for (int off = 128; off > 0; off >>= 1) {
        if (tid < off) red[tid] += red[tid + off];
        __syncthreads();
    }
    float mean = red[0] * (1.0f / Dc);
    __syncthreads();
    float vs = 0.f;
    for (int i = tid; i < Dc; i += 256) { float d = xr[i] - mean; vs += d * d; }
    red[tid] = vs; __syncthreads();
    for (int off = 128; off > 0; off >>= 1) {
        if (tid < off) red[tid] += red[tid + off];
        __syncthreads();
    }
    float rstd = rsqrtf(red[0] * (1.0f / Dc) + 1e-5f);
    for (int i = tid; i < Dc; i += 256)
        out[(size_t)t * Dc + i] = (__bf16)((xr[i] - mean) * rstd * g[i] + b[i]);
}

// ---------------- generic WMMA bf16 GEMM (async double-buffered) ---------
// C[M,N] = A[M,K] @ Bt[N,K]^T   (Bt stored transposed, K-contiguous)
// MODE 0: Cf = acc + bias                     (f32 out)
// MODE 1: Cf = resid + acc + bias             (f32 out)
// MODE 2: Cb = bf16(gelu(acc + bias))         (bf16 out)
// MODE 3: Cf[row] += scale*rowscale[m]*(acc+bias)  (scatter-accumulate)
template<int MODE>
__global__ __launch_bounds__(256)
void gemm_wmma(const __bf16* __restrict__ A, const __bf16* __restrict__ Bt,
               const float* __restrict__ bias,
               float* __restrict__ Cf, __bf16* __restrict__ Cb,
               const float* __restrict__ resid,
               const int* __restrict__ arowidx, const int* __restrict__ crowidx,
               const float* __restrict__ rowscale, const int* __restrict__ cnt,
               float scale, int M, int N, int K)
{
    __shared__ __align__(64) __bf16 sA[2][128 * 32];
    __shared__ __align__(64) __bf16 sB[2][128 * 32];

    int Mc = cnt ? min(M, *cnt) : M;
    int m0 = blockIdx.y * 128;
    int n0 = blockIdx.x * 128;
    if (m0 >= Mc) return;

    int tid  = threadIdx.x;
    int lane = tid & 31, wave = tid >> 5;
    int wm   = (wave >> 1) * 32;     // 4 waves along M
    int wn   = (wave & 1) * 64;      // 2 waves along N
    int half = lane >> 4, l15 = lane & 15;
    int off0 = half * 8;

    v8f acc[2][4] = {};

    // stage one 128x32 A tile + 128x32 B tile into buffer `buf` (async)
    auto stage = [&](int k0, int buf) {
        #pragma unroll
        for (int c0 = 0; c0 < 2; c0++) {
            int c = tid + c0 * 256;               // 512 chunks of 8 bf16
            int row = c >> 2, seg = (c & 3) * 8;
            int gm = m0 + row;
            int ar = gm < Mc ? gm : (Mc - 1);
            if (arowidx) ar = arowidx[ar];
            async_load16(&A[(size_t)ar * K + k0 + seg], &sA[buf][row * 32 + seg]);
        }
        #pragma unroll
        for (int c0 = 0; c0 < 2; c0++) {
            int c = tid + c0 * 256;
            int row = c >> 2, seg = (c & 3) * 8;
            async_load16(&Bt[(size_t)(n0 + row) * K + k0 + seg],
                         &sB[buf][row * 32 + seg]);
        }
    };

    int nk = K >> 5;
    stage(0, 0);
    for (int it = 0; it < nk; it++) {
        int cur = it & 1;
        wait_async0();          // this wave's async writes to LDS are done
        __syncthreads();        // ... and everyone else's: tile `it` visible
        if (it + 1 < nk)        // overlap next tile's copy with compute
            stage((it + 1) << 5, 1 - cur);

        FragA a[2]; FragB b[4];
        #pragma unroll
        for (int tm = 0; tm < 2; tm++) {
            int mrow = wm + tm * 16 + l15;
            a[tm].h[0] = *(const v8bf*)&sA[cur][mrow * 32 + off0];
            a[tm].h[1] = *(const v8bf*)&sA[cur][mrow * 32 + off0 + 16];
        }
        #pragma unroll
        for (int tn = 0; tn < 4; tn++) {
            int nrow = wn + tn * 16 + l15;
            b[tn].h[0] = *(const v8bf*)&sB[cur][nrow * 32 + off0];
            b[tn].h[1] = *(const v8bf*)&sB[cur][nrow * 32 + off0 + 16];
        }
        #pragma unroll
        for (int tm = 0; tm < 2; tm++)
            #pragma unroll
            for (int tn = 0; tn < 4; tn++)
                acc[tm][tn] = __builtin_amdgcn_wmma_f32_16x16x32_bf16(
                    false, a[tm].v, false, b[tn].v,
                    (short)0, acc[tm][tn], false, false);
    }

    // epilogue: C layout M = r + 8*half, N = l15
    #pragma unroll
    for (int tm = 0; tm < 2; tm++)
     #pragma unroll
     for (int tn = 0; tn < 4; tn++)
      #pragma unroll
      for (int r = 0; r < 8; r++) {
          int m = m0 + wm + tm * 16 + r + half * 8;
          int n = n0 + wn + tn * 16 + l15;
          if (m >= Mc) continue;
          float vo = acc[tm][tn][r] + (bias ? bias[n] : 0.0f);
          if (MODE == 0) {
              Cf[(size_t)m * N + n] = vo;
          } else if (MODE == 1) {
              Cf[(size_t)m * N + n] = resid[(size_t)m * N + n] + vo;
          } else if (MODE == 2) {
              Cb[(size_t)m * N + n] = (__bf16)gelu_f(vo);
          } else {
              int row  = crowidx ? crowidx[m] : m;
              float sc = scale * (rowscale ? rowscale[m] : 1.0f);
              Cf[(size_t)row * N + n] += sc * vo;
          }
      }
}

// ---------------- RoPE + latent compression ----------------
__global__ __launch_bounds__(256)
void qkv_post(const float* __restrict__ q, const float* __restrict__ k,
              const float* __restrict__ v,
              const float* __restrict__ cosT, const float* __restrict__ sinT,
              const float* __restrict__ Wqc, const float* __restrict__ bqc,
              const float* __restrict__ Wkc, const float* __restrict__ bkc,
              const float* __restrict__ Wvc, const float* __restrict__ bvc,
              __bf16* __restrict__ qcb, __bf16* __restrict__ kcb,
              __bf16* __restrict__ vcT)
{
    int idx = blockIdx.x * blockDim.x + threadIdx.x;
    if (idx >= Tc * NHc) return;
    int t = idx / NHc, h = idx % NHc;
    int b = t / Lc, l = t % Lc;

    const float* qp = q + (size_t)t * Dc + h * HDc;
    const float* kp = k + (size_t)t * Dc + h * HDc;
    const float* vp = v + (size_t)t * Dc + h * HDc;

    float qr[HDc], kr[HDc], vv[HDc];
    #pragma unroll 4
    for (int i = 0; i < HDc / 2; i++) {
        float c = cosT[l * 32 + i], s = sinT[l * 32 + i];
        float qe = qp[2 * i], qo = qp[2 * i + 1];
        float ke = kp[2 * i], ko = kp[2 * i + 1];
        qr[2 * i] = qe * c - qo * s; qr[2 * i + 1] = qe * s + qo * c;
        kr[2 * i] = ke * c - ko * s; kr[2 * i + 1] = ke * s + ko * c;
        vv[2 * i] = vp[2 * i];       vv[2 * i + 1] = vp[2 * i + 1];
    }
    size_t bh = (size_t)b * NHc + h;
    size_t rowbase = (bh * Lc + l) * 32;     // K padded 16->32
    for (int c = 0; c < LATc; c++) {
        float aq = bqc[c], ak = bkc[c], av = bvc[c];
        #pragma unroll 8
        for (int d = 0; d < HDc; d++) {
            aq += qr[d] * Wqc[d * LATc + c];
            ak += kr[d] * Wkc[d * LATc + c];
            av += vv[d] * Wvc[d * LATc + c];
        }
        qcb[rowbase + c] = (__bf16)aq;
        kcb[rowbase + c] = (__bf16)ak;
        vcT[(bh * LATc + c) * Lc + l] = (__bf16)av;     // transposed for B-frag
    }
    for (int c = LATc; c < 32; c++) {
        qcb[rowbase + c] = (__bf16)0.f;
        kcb[rowbase + c] = (__bf16)0.f;
    }
}

// ---------------- flash attention (latent dim 16, padded to K=32) --------
__global__ __launch_bounds__(32)
void flash_attn(const __bf16* __restrict__ qcb, const __bf16* __restrict__ kcb,
                const __bf16* __restrict__ vcT, float* __restrict__ attnc)
{
    int nqb = Lc / 16;
    int blk = blockIdx.x;
    int bh  = blk / nqb, qb = blk % nqb;
    int lane = threadIdx.x, half = lane >> 4, l15 = lane & 15;
    int off0 = half * 8;

    __shared__ __align__(64) __bf16 sP[16 * 32];
    for (int i = lane; i < 16 * 32; i += 32) sP[i] = (__bf16)0.f;

    FragA aq;
    size_t qrow = ((size_t)bh * Lc + qb * 16 + l15) * 32;
    aq.h[0] = *(const v8bf*)&qcb[qrow + off0];
    aq.h[1] = *(const v8bf*)&qcb[qrow + off0 + 16];

    float rowm[8], rowl[8];
    v8f o = {};
    #pragma unroll
    for (int r = 0; r < 8; r++) { rowm[r] = -1e30f; rowl[r] = 0.f; }
    __syncthreads();

    for (int kb = 0; kb <= qb; kb++) {
        FragB bk;
        size_t krow = ((size_t)bh * Lc + kb * 16 + l15) * 32;
        bk.h[0] = *(const v8bf*)&kcb[krow + off0];
        bk.h[1] = *(const v8bf*)&kcb[krow + off0 + 16];

        v8f zc = {};
        v8f s = __builtin_amdgcn_wmma_f32_16x16x32_bf16(
            false, aq.v, false, bk.v, (short)0, zc, false, false);

        float p[8];
        #pragma unroll
        for (int r = 0; r < 8; r++) {
            float sv = s[r] * 0.25f;                 // 1/sqrt(LAT)
            int mloc = r + half * 8;
            int qidx = qb * 16 + mloc, kidx = kb * 16 + l15;
            if (kidx > qidx) sv = -1e30f;            // causal
            float mr = sv;
            for (int msk = 1; msk < 16; msk <<= 1)
                mr = fmaxf(mr, __shfl_xor(mr, msk, 32));
            float mnew = fmaxf(rowm[r], mr);
            float pe = __expf(sv - mnew);
            float ps = pe;
            for (int msk = 1; msk < 16; msk <<= 1)
                ps += __shfl_xor(ps, msk, 32);
            float corr = __expf(rowm[r] - mnew);
            rowl[r] = rowl[r] * corr + ps;
            rowm[r] = mnew;
            o[r] *= corr;
            p[r] = pe;
        }
        __syncthreads();
        #pragma unroll
        for (int r = 0; r < 8; r++)
            sP[(r + half * 8) * 32 + l15] = (__bf16)p[r];
        __syncthreads();

        FragA ap;
        ap.v = *(const v16bf*)&sP[l15 * 32 + half * 16];
        FragB bv;
        if (half == 0) {
            bv.v = *(const v16bf*)&vcT[((size_t)bh * LATc + l15) * Lc + kb * 16];
        } else {
            #pragma unroll
            for (int i = 0; i < 16; i++) ((__bf16*)&bv)[i] = (__bf16)0.f;  // K pad
        }
        o = __builtin_amdgcn_wmma_f32_16x16x32_bf16(
            false, ap.v, false, bv.v, (short)0, o, false, false);
    }

    #pragma unroll
    for (int r = 0; r < 8; r++) {
        int mloc = r + half * 8;
        attnc[((size_t)bh * Lc + qb * 16 + mloc) * LATc + l15] = o[r] / rowl[r];
    }
}

// ---------------- attn @ Wd + bd -> bf16 [T,D] ----------------
__global__ __launch_bounds__(256)
void wd_kernel(const float* __restrict__ attnc, const float* __restrict__ Wd,
               const float* __restrict__ bd, __bf16* __restrict__ attnb)
{
    int i = blockIdx.x * blockDim.x + threadIdx.x;
    if (i >= Tc * Dc) return;
    int t = i >> 10, col = i & 1023;
    int h = col >> 6, d = col & 63;
    int b = t / Lc, l = t % Lc;
    const float* ac = attnc + (((size_t)b * NHc + h) * Lc + l) * LATc;
    float s = bd[d];
    #pragma unroll
    for (int c = 0; c < LATc; c++) s += ac[c] * Wd[c * HDc + d];
    attnb[i] = (__bf16)s;
}

// ---------------- weight convert + transpose f32[K,N] -> bf16[N,K] -------
__global__ __launch_bounds__(256)
void convT(const float* __restrict__ W, __bf16* __restrict__ Wt, int K, int N)
{
    size_t i = (size_t)blockIdx.x * 256 + threadIdx.x;
    if (i >= (size_t)K * N) return;
    int kk = (int)(i / N), n = (int)(i % N);
    Wt[(size_t)n * K + kk] = (__bf16)W[i];
}

// ---------------- zero accumulator + expert counters ----------------
__global__ __launch_bounds__(256)
void init_zero(float* __restrict__ ffnacc, int* __restrict__ cnt)
{
    int i = blockIdx.x * blockDim.x + threadIdx.x;
    if (i < Tc * Dc) ffnacc[i] = 0.f;
    if (i < Ec) cnt[i] = 0;
}

// ---------------- router: top-2 softmax + token scatter ----------------
__global__ __launch_bounds__(256)
void gates_kernel(const __bf16* __restrict__ h2b, const float* __restrict__ Wr,
                  const float* __restrict__ br, int* __restrict__ cnt,
                  int* __restrict__ eidx, float* __restrict__ ew)
{
    int t = blockIdx.x * blockDim.x + threadIdx.x;
    if (t >= Tc) return;
    float acc[Ec];
    #pragma unroll
    for (int e = 0; e < Ec; e++) acc[e] = br[e];
    const __bf16* hr = h2b + (size_t)t * Dc;
    for (int kk = 0; kk < Dc; kk++) {
        float hv = (float)hr[kk];
        #pragma unroll
        for (int e = 0; e < Ec; e++) acc[e] += hv * Wr[kk * Ec + e];
    }
    int e1 = 0; float g1 = acc[0];
    #pragma unroll
    for (int e = 1; e < Ec; e++) if (acc[e] > g1) { g1 = acc[e]; e1 = e; }
    int e2 = -1; float g2 = -1e30f;
    #pragma unroll
    for (int e = 0; e < Ec; e++)
        if (e != e1 && acc[e] > g2) { g2 = acc[e]; e2 = e; }
    float w1 = 1.0f / (1.0f + __expf(g2 - g1));
    float w2 = 1.0f - w1;
    int p1 = atomicAdd(&cnt[e1], 1); eidx[e1 * Tc + p1] = t; ew[e1 * Tc + p1] = w1;
    int p2 = atomicAdd(&cnt[e2], 1); eidx[e2 * Tc + p2] = t; ew[e2 * Tc + p2] = w2;
}

// ---------------- final residual add ----------------
__global__ __launch_bounds__(256)
void final_add(const float* __restrict__ x1, const float* __restrict__ ffnacc,
               float* __restrict__ out)
{
    int i = blockIdx.x * blockDim.x + threadIdx.x;
    if (i < Tc * Dc) out[i] = x1[i] + ffnacc[i];
}

// =====================================================================
extern "C" void kernel_launch(void* const* d_in, const int* in_sizes, int n_in,
                              void* d_out, int out_size, void* d_ws, size_t ws_size,
                              hipStream_t stream)
{
    (void)in_sizes; (void)n_in; (void)out_size; (void)ws_size;
    const float* x     = (const float*)d_in[0];
    const float* cosT  = (const float*)d_in[1];
    const float* sinT  = (const float*)d_in[2];
    const float* ln1_g = (const float*)d_in[3];
    const float* ln1_b = (const float*)d_in[4];
    const float* Wq    = (const float*)d_in[5];
    const float* bq    = (const float*)d_in[6];
    const float* Wk    = (const float*)d_in[7];
    const float* bk    = (const float*)d_in[8];
    const float* Wv    = (const float*)d_in[9];
    const float* bv    = (const float*)d_in[10];
    const float* Wqc   = (const float*)d_in[11];
    const float* bqc   = (const float*)d_in[12];
    const float* Wkc   = (const float*)d_in[13];
    const float* bkc   = (const float*)d_in[14];
    const float* Wvc   = (const float*)d_in[15];
    const float* bvc   = (const float*)d_in[16];
    const float* Wd    = (const float*)d_in[17];
    const float* bd    = (const float*)d_in[18];
    const float* Wo    = (const float*)d_in[19];
    const float* bo    = (const float*)d_in[20];
    const float* ln2_g = (const float*)d_in[21];
    const float* ln2_b = (const float*)d_in[22];
    const float* Wr    = (const float*)d_in[23];
    const float* br    = (const float*)d_in[24];
    const float* eW1   = (const float*)d_in[25];
    const float* eb1   = (const float*)d_in[26];
    const float* eW2   = (const float*)d_in[27];
    const float* eb2   = (const float*)d_in[28];
    const float* sW1   = (const float*)d_in[29];
    const float* sb1   = (const float*)d_in[30];
    const float* sW2   = (const float*)d_in[31];
    const float* sb2   = (const float*)d_in[32];

    // -------- workspace carve-up --------
    char* ws = (char*)d_ws;
    size_t off = 0;
    auto alloc = [&](size_t bytes) -> char* {
        char* p = ws + off;
        off += (bytes + 255) & ~(size_t)255;
        return p;
    };
    __bf16* hbf    = (__bf16*)alloc((size_t)Tc * Dc * 2);
    float*  qf     = (float*) alloc((size_t)Tc * Dc * 4);
    float*  kf     = (float*) alloc((size_t)Tc * Dc * 4);
    float*  vf     = (float*) alloc((size_t)Tc * Dc * 4);
    __bf16* qcb    = (__bf16*)alloc((size_t)Bc * NHc * Lc * 32 * 2);
    __bf16* kcb    = (__bf16*)alloc((size_t)Bc * NHc * Lc * 32 * 2);
    __bf16* vcT    = (__bf16*)alloc((size_t)Bc * NHc * LATc * Lc * 2);
    float*  attnc  = (float*) alloc((size_t)Bc * NHc * Lc * LATc * 4);
    __bf16* attnb  = (__bf16*)alloc((size_t)Tc * Dc * 2);
    float*  x1     = (float*) alloc((size_t)Tc * Dc * 4);
    __bf16* h2b    = (__bf16*)alloc((size_t)Tc * Dc * 2);
    __bf16* act    = (__bf16*)alloc((size_t)Tc * HFFc * 2);
    float*  ffnacc = (float*) alloc((size_t)Tc * Dc * 4);
    int*    cnt    = (int*)   alloc(Ec * 4);
    int*    eidx   = (int*)   alloc((size_t)Ec * Tc * 4);
    float*  ewgt   = (float*) alloc((size_t)Ec * Tc * 4);
    __bf16* Wqt    = (__bf16*)alloc((size_t)Dc * Dc * 2);
    __bf16* Wkt    = (__bf16*)alloc((size_t)Dc * Dc * 2);
    __bf16* Wvt    = (__bf16*)alloc((size_t)Dc * Dc * 2);
    __bf16* Wot    = (__bf16*)alloc((size_t)Dc * Dc * 2);
    __bf16* w1t    = (__bf16*)alloc((size_t)Dc * HFFc * 2);   // reused per expert
    __bf16* w2t    = (__bf16*)alloc((size_t)HFFc * Dc * 2);

    const int EL = 256;
    auto blks = [](size_t n, int bs) { return (unsigned)((n + bs - 1) / bs); };

    // weights (attention path) -> bf16 transposed
    convT<<<blks((size_t)Dc * Dc, EL), EL, 0, stream>>>(Wq, Wqt, Dc, Dc);
    convT<<<blks((size_t)Dc * Dc, EL), EL, 0, stream>>>(Wk, Wkt, Dc, Dc);
    convT<<<blks((size_t)Dc * Dc, EL), EL, 0, stream>>>(Wv, Wvt, Dc, Dc);
    convT<<<blks((size_t)Dc * Dc, EL), EL, 0, stream>>>(Wo, Wot, Dc, Dc);

    // LN1
    ln_kernel<<<Tc, 256, 0, stream>>>(x, ln1_g, ln1_b, hbf);

    // Q/K/V projections (WMMA)
    dim3 gQKV(Dc / 128, Tc / 128);
    gemm_wmma<0><<<gQKV, 256, 0, stream>>>(hbf, Wqt, bq, qf, nullptr, nullptr,
        nullptr, nullptr, nullptr, nullptr, 1.f, Tc, Dc, Dc);
    gemm_wmma<0><<<gQKV, 256, 0, stream>>>(hbf, Wkt, bk, kf, nullptr, nullptr,
        nullptr, nullptr, nullptr, nullptr, 1.f, Tc, Dc, Dc);
    gemm_wmma<0><<<gQKV, 256, 0, stream>>>(hbf, Wvt, bv, vf, nullptr, nullptr,
        nullptr, nullptr, nullptr, nullptr, 1.f, Tc, Dc, Dc);

    // RoPE + latent compression
    qkv_post<<<blks((size_t)Tc * NHc, EL), EL, 0, stream>>>(
        qf, kf, vf, cosT, sinT, Wqc, bqc, Wkc, bkc, Wvc, bvc, qcb, kcb, vcT);

    // flash attention over latents
    flash_attn<<<Bc * NHc * (Lc / 16), 32, 0, stream>>>(qcb, kcb, vcT, attnc);

    // decompress heads: @Wd + bd
    wd_kernel<<<blks((size_t)Tc * Dc, EL), EL, 0, stream>>>(attnc, Wd, bd, attnb);

    // output projection + residual
    gemm_wmma<1><<<gQKV, 256, 0, stream>>>(attnb, Wot, bo, x1, nullptr, x,
        nullptr, nullptr, nullptr, nullptr, 1.f, Tc, Dc, Dc);

    // LN2
    ln_kernel<<<Tc, 256, 0, stream>>>(x1, ln2_g, ln2_b, h2b);

    // zero FFN accumulator + expert counters, then route
    init_zero<<<blks((size_t)Tc * Dc, EL), EL, 0, stream>>>(ffnacc, cnt);
    gates_kernel<<<blks(Tc, EL), EL, 0, stream>>>(h2b, Wr, br, cnt, eidx, ewgt);

    dim3 gFF1(HFFc / 128, Tc / 128);
    dim3 gFF2(Dc / 128, Tc / 128);

    // shared experts (dense), averaged by 1/S
    for (int s = 0; s < Sc; s++) {
        convT<<<blks((size_t)Dc * HFFc, EL), EL, 0, stream>>>(
            sW1 + (size_t)s * Dc * HFFc, w1t, Dc, HFFc);
        gemm_wmma<2><<<gFF1, 256, 0, stream>>>(h2b, w1t, sb1 + (size_t)s * HFFc,
            nullptr, act, nullptr, nullptr, nullptr, nullptr, nullptr,
            1.f, Tc, HFFc, Dc);
        convT<<<blks((size_t)HFFc * Dc, EL), EL, 0, stream>>>(
            sW2 + (size_t)s * HFFc * Dc, w2t, HFFc, Dc);
        gemm_wmma<3><<<gFF2, 256, 0, stream>>>(act, w2t, sb2 + (size_t)s * Dc,
            ffnacc, nullptr, nullptr, nullptr, nullptr, nullptr, nullptr,
            1.0f / Sc, Tc, Dc, HFFc);
    }

    // routed experts (sparse: gather rows -> GEMM -> scatter-scale)
    for (int e = 0; e < Ec; e++) {
        convT<<<blks((size_t)Dc * HFFc, EL), EL, 0, stream>>>(
            eW1 + (size_t)e * Dc * HFFc, w1t, Dc, HFFc);
        gemm_wmma<2><<<gFF1, 256, 0, stream>>>(h2b, w1t, eb1 + (size_t)e * HFFc,
            nullptr, act, nullptr, eidx + (size_t)e * Tc, nullptr, nullptr,
            cnt + e, 1.f, Tc, HFFc, Dc);
        convT<<<blks((size_t)HFFc * Dc, EL), EL, 0, stream>>>(
            eW2 + (size_t)e * HFFc * Dc, w2t, HFFc, Dc);
        gemm_wmma<3><<<gFF2, 256, 0, stream>>>(act, w2t, eb2 + (size_t)e * Dc,
            ffnacc, nullptr, nullptr, nullptr, eidx + (size_t)e * Tc,
            ewgt + (size_t)e * Tc, cnt + e, 1.f, Tc, Dc, HFFc);
    }

    // final residual
    final_add<<<blks((size_t)Tc * Dc, EL), EL, 0, stream>>>(x1, ffnacc, (float*)d_out);
}
